// SSMWorld_37014028157383
// MI455X (gfx1250) — matrix operand order repaired and checked by reading, not compiled, not run on hardware
//
#include <hip/hip_runtime.h>

// ---------------------------------------------------------------------------
// SSM layer for MI455X (gfx1250): f16 WMMA GEMMs with async-to-LDS double
// buffering + batched WMMA recurrence.  B=128, T=256, H=512, S=256.
// ---------------------------------------------------------------------------

typedef __attribute__((ext_vector_type(16))) _Float16 v16h;
typedef __attribute__((ext_vector_type(8)))  float    v8f;

#define HID   512
#define ST    256
#define BSZ   128
#define TSZ   256
#define MROWS (BSZ * TSZ)   // 32768 rows of x / Bx / hs / out

// ----- CDNA5 async global->LDS copy support (probe via __has_builtin) -------
#if defined(__has_builtin)
#if __has_builtin(__builtin_amdgcn_global_load_async_to_lds_b128)
#define HAVE_ASYNC_LDS 1
#endif
#endif

#ifdef HAVE_ASYNC_LDS
typedef int async16_t __attribute__((vector_size(16)));   // matches builtin param
#if __has_builtin(__builtin_amdgcn_s_wait_asynccnt)
#define WAIT_ASYNC(n) __builtin_amdgcn_s_wait_asynccnt(n)
#else
#define WAIT_ASYNC(n) asm volatile("s_wait_asynccnt %0" ::"i"(n) : "memory")
#endif
#define ASYNC_CP16(gsrc, ldst)                                                 \
  __builtin_amdgcn_global_load_async_to_lds_b128(                              \
      (async16_t __attribute__((address_space(1)))*)(gsrc),                    \
      (async16_t __attribute__((address_space(3)))*)(ldst), 0, 0)
#else
#define WAIT_ASYNC(n)
#endif

__device__ __forceinline__ v8f wmma16(v16h a, v16h b, v8f c) {
  return __builtin_amdgcn_wmma_f32_16x16x32_f16(
      /*neg_a=*/false, a, /*neg_b=*/false, b,
      /*c_mod=*/(short)0, c, /*reuse_a=*/false, /*reuse_b=*/false);
}

// A-fragment: 16x32 f16 tile, row-major, stride lda (contiguous k: lane half h holds k = 16h..16h+15)
__device__ __forceinline__ v16h load_a_frag(const _Float16* base, int lda, int lane) {
  int ar    = lane & 15;
  int khalf = (lane >> 4) & 1;
  v16h a;
#pragma unroll
  for (int v = 0; v < 8; ++v) {
    int kk = (khalf << 4) + (v << 1);
    a[2 * v]     = base[ar * lda + kk];
    a[2 * v + 1] = base[ar * lda + kk + 1];
  }
  return a;
}

// B-fragment: 32x16 f16 tile (K x N), row-major, stride ldb
__device__ __forceinline__ v16h load_b_frag(const _Float16* base, int ldb, int lane) {
  int bn = lane & 15;
  int kh = ((lane >> 4) & 1) << 4;
  v16h b;
#pragma unroll
  for (int v = 0; v < 8; ++v) {
    int kk = kh + 2 * v;
    b[2 * v]     = base[kk * ldb + bn];
    b[2 * v + 1] = base[(kk + 1) * ldb + bn];
  }
  return b;
}

// ---------------------------------------------------------------------------
// All-f16 WMMA GEMM:  C[M,N] (fp32) = A[M,K] @ Bt[K,N] + bias[N]
//   A  : f16 row-major [M,K]      Bt : f16 row-major [K,N] (pre-transposed W)
//   ACCUM: C += result; ADD_RES: add fp32 residual Res[M,N]
// Tile 128x64x32, 256 threads = 8 waves (4x2), 2x2 WMMA per wave.
// LDS double-buffered; staged with GLOBAL_LOAD_ASYNC_TO_LDS_B128 when present.
// Per-thread async issues per tile: 2 (As) + 1 (Bs) = 3.
// ---------------------------------------------------------------------------
#define AS_LD 40   // 32 + 8 pad -> 80B row stride (16B multiple)
#define BS_LD 72   // 64 + 8 pad -> 144B row stride (16B multiple)

__device__ __forceinline__ void stage_tile(const _Float16* __restrict__ A,
                                           const _Float16* __restrict__ Bt,
                                           _Float16 (&As)[BSZ][AS_LD],
                                           _Float16 (&Bs)[32][BS_LD],
                                           int bm0, int bn0, int k0,
                                           int N, int K, int tid) {
#pragma unroll
  for (int i = 0; i < 2; ++i) {          // A tile: 128 rows x 64B = 512 chunks
    int idx = tid + i * 256;
    int r = idx >> 2, c4 = idx & 3;
    const _Float16* src = A + (size_t)(bm0 + r) * K + k0 + c4 * 8;
#ifdef HAVE_ASYNC_LDS
    ASYNC_CP16(src, &As[r][c4 * 8]);
#else
    *(int4*)&As[r][c4 * 8] = *(const int4*)src;
#endif
  }
  {                                      // B tile: 32 rows x 128B = 256 chunks
    int r = tid >> 3, c8 = tid & 7;
    const _Float16* src = Bt + (size_t)(k0 + r) * N + bn0 + c8 * 8;
#ifdef HAVE_ASYNC_LDS
    ASYNC_CP16(src, &Bs[r][c8 * 8]);
#else
    *(int4*)&Bs[r][c8 * 8] = *(const int4*)src;
#endif
  }
}

template <bool ACCUM, bool ADD_RES>
__global__ __launch_bounds__(256)
void gemm_wmma_f16(const _Float16* __restrict__ A, const _Float16* __restrict__ Bt,
                   const float* __restrict__ bias, const float* __restrict__ Res,
                   float* __restrict__ C, int M, int N, int K) {
  __shared__ __align__(16) _Float16 As[2][BSZ][AS_LD];
  __shared__ __align__(16) _Float16 Bs[2][32][BS_LD];

  int tid = threadIdx.x, lane = tid & 31, wid = tid >> 5;
  int wm = wid & 3;          // wave row block (32 rows)
  int wn = wid >> 2;         // wave col block (32 cols)
  int bm0 = blockIdx.y * BSZ;
  int bn0 = blockIdx.x * 64;

  v8f zero8 = {0.f, 0.f, 0.f, 0.f, 0.f, 0.f, 0.f, 0.f};
  v8f acc00 = zero8, acc01 = zero8, acc10 = zero8, acc11 = zero8;

  int nk = K >> 5;
  stage_tile(A, Bt, As[0], Bs[0], bm0, bn0, 0, N, K, tid);   // prologue

  for (int kc = 0; kc < nk; ++kc) {
    int cur = kc & 1;
    if (kc + 1 < nk) {       // prefetch next tile into the other buffer
      stage_tile(A, Bt, As[1 - cur], Bs[1 - cur], bm0, bn0, (kc + 1) * 32, N, K, tid);
      WAIT_ASYNC(3);         // current tile's 3 async copies complete
    } else {
      WAIT_ASYNC(0);
    }
    __syncthreads();

    v16h a0 = load_a_frag(&As[cur][wm * 32][0],      AS_LD, lane);
    v16h a1 = load_a_frag(&As[cur][wm * 32 + 16][0], AS_LD, lane);
    v16h b0 = load_b_frag(&Bs[cur][0][wn * 32],      BS_LD, lane);
    v16h b1 = load_b_frag(&Bs[cur][0][wn * 32 + 16], BS_LD, lane);

    acc00 = wmma16(a0, b0, acc00);
    acc01 = wmma16(a0, b1, acc01);
    acc10 = wmma16(a1, b0, acc10);
    acc11 = wmma16(a1, b1, acc11);
    __syncthreads();
  }

  int mhi = ((lane >> 4) & 1) * 8;
  int nn  = lane & 15;
#pragma unroll
  for (int mi = 0; mi < 2; ++mi) {
#pragma unroll
    for (int ni = 0; ni < 2; ++ni) {
      v8f av = (mi == 0) ? ((ni == 0) ? acc00 : acc01)
                         : ((ni == 0) ? acc10 : acc11);
#pragma unroll
      for (int v = 0; v < 8; ++v) {
        int gm = bm0 + wm * 32 + mi * 16 + v + mhi;
        int gn = bn0 + wn * 32 + ni * 16 + nn;
        float val = av[v] + bias[gn];
        if (ADD_RES) val += Res[(size_t)gm * N + gn];
        if (ACCUM)   val += C[(size_t)gm * N + gn];
        C[(size_t)gm * N + gn] = val;
      }
    }
  }
}

// ---------------------------------------------------------------------------
// Batched linear recurrence:  h_t = h_{t-1} @ Ad^T + Bx_t
// 8 blocks x 512 threads; block = 16 batches; wave owns one 16-wide N slice;
// Ad^T fragments preloaded into VGPRs once (constant over t); h in LDS (f16).
// ---------------------------------------------------------------------------
__global__ __launch_bounds__(512)
void ssm_scan(const float* __restrict__ Bx,      // [B,T,S] fp32
              const _Float16* __restrict__ AdT,  // [S,S] f16: AdT[k][n]=Ad[n][k]
              _Float16* __restrict__ hs) {       // [B,T,S] f16 out
  __shared__ _Float16 hsm[16][ST + 8];
  int tid = threadIdx.x, lane = tid & 31, wid = tid >> 5;
  int b0 = blockIdx.x << 4;
  int n0 = wid << 4;

  int bn = lane & 15;
  int kh = ((lane >> 4) & 1) << 4;
  v16h Bf[8];
#pragma unroll
  for (int kc = 0; kc < 8; ++kc) {
#pragma unroll
    for (int v = 0; v < 8; ++v) {
      int kk = kc * 32 + kh + 2 * v;
      Bf[kc][2 * v]     = AdT[(size_t)kk * ST + n0 + bn];
      Bf[kc][2 * v + 1] = AdT[(size_t)(kk + 1) * ST + n0 + bn];
    }
  }

  for (int i = tid; i < 16 * (ST + 8); i += 512)
    hsm[i / (ST + 8)][i % (ST + 8)] = (_Float16)0.f;
  __syncthreads();

  int mhi = ((lane >> 4) & 1) * 8;
  int nn  = lane & 15;
  for (int t = 0; t < TSZ; ++t) {
    v8f acc;
#pragma unroll
    for (int v = 0; v < 8; ++v)
      acc[v] = Bx[((size_t)(b0 + v + mhi) * TSZ + t) * ST + n0 + nn];

#pragma unroll
    for (int kc = 0; kc < 8; ++kc) {
      v16h a = load_a_frag(&hsm[0][kc * 32], ST + 8, lane);
      acc = wmma16(a, Bf[kc], acc);
    }
    __syncthreads();

#pragma unroll
    for (int v = 0; v < 8; ++v) {
      int m = v + mhi;
      _Float16 hv = (_Float16)acc[v];
      hsm[m][n0 + nn] = hv;
      hs[((size_t)(b0 + m) * TSZ + t) * ST + n0 + nn] = hv;
    }
    __syncthreads();
  }
}

// ---------------------------------------------------------------------------
// LayerNorm over H=512, in place. One 256-thread block per row.
// ---------------------------------------------------------------------------
__global__ __launch_bounds__(256)
void layernorm_512(float* __restrict__ out, const float* __restrict__ gamma,
                   const float* __restrict__ beta) {
  __shared__ float s1[256], s2[256];
  size_t row = blockIdx.x;
  int tid = threadIdx.x;
  float v0 = out[row * HID + tid];
  float v1 = out[row * HID + tid + 256];
  s1[tid] = v0 + v1;
  s2[tid] = v0 * v0 + v1 * v1;
  __syncthreads();
  for (int off = 128; off > 0; off >>= 1) {
    if (tid < off) { s1[tid] += s1[tid + off]; s2[tid] += s2[tid + off]; }
    __syncthreads();
  }
  float mu  = s1[0] * (1.f / HID);
  float var = s2[0] * (1.f / HID) - mu * mu;
  float inv = rsqrtf(var + 1e-5f);
  out[row * HID + tid]       = (v0 - mu) * inv * gamma[tid] + beta[tid];
  out[row * HID + tid + 256] = (v1 - mu) * inv * gamma[tid + 256] + beta[tid + 256];
}

// ---------------------------------------------------------------------------
// expm(A*DT): scale 1/16 + Taylor(12) + 4 squarings (tiny fp32 job).
// ---------------------------------------------------------------------------
__global__ __launch_bounds__(256)
void expm_prep(const float* __restrict__ A, float* __restrict__ M,
               float* __restrict__ T0, float* __restrict__ P, float scale) {
  int id = blockIdx.x * 256 + threadIdx.x;
  int i = id >> 8, j = id & 255;
  float m = A[id] * scale;
  M[id] = m; T0[id] = m;
  P[id] = m + (i == j ? 1.f : 0.f);
}

__global__ __launch_bounds__(256)
void expm_mm(const float* __restrict__ X, const float* __restrict__ Y,
             float* __restrict__ Z, float* __restrict__ P, float coef) {
  int id = blockIdx.x * 256 + threadIdx.x;
  int i = id >> 8, j = id & 255;
  float s = 0.f;
  for (int k = 0; k < ST; ++k)
    s = fmaf(X[i * ST + k], Y[k * ST + j], s);
  s *= coef;
  Z[id] = s;
  if (P) P[id] += s;
}

__global__ __launch_bounds__(256)
void expm_transpose_f16(const float* __restrict__ P, _Float16* __restrict__ AdT) {
  int id = blockIdx.x * 256 + threadIdx.x;
  int k = id >> 8, n = id & 255;
  AdT[id] = (_Float16)P[n * ST + k];
}

// ---------------------------------------------------------------------------
// One-time conversions: x -> f16; weights -> transposed f16 [K,N].
// ---------------------------------------------------------------------------
__global__ __launch_bounds__(256)
void cvt_f16(const float* __restrict__ src, _Float16* __restrict__ dst) {
  size_t id = (size_t)blockIdx.x * 256 + threadIdx.x;
  dst[id] = (_Float16)src[id];
}

__global__ __launch_bounds__(256)
void transpose_f16(const float* __restrict__ W, _Float16* __restrict__ Wt,
                   int N, int K) {  // W:[N,K] fp32 -> Wt:[K,N] f16
  int id = blockIdx.x * 256 + threadIdx.x;
  int k = id / N, n = id % N;
  Wt[id] = (_Float16)W[(size_t)n * K + k];
}

// ---------------------------------------------------------------------------
// Host-side orchestration
// ---------------------------------------------------------------------------
extern "C" void kernel_launch(void* const* d_in, const int* in_sizes, int n_in,
                              void* d_out, int out_size, void* d_ws, size_t ws_size,
                              hipStream_t stream) {
  (void)in_sizes; (void)n_in; (void)out_size; (void)ws_size;
  const float* x     = (const float*)d_in[0];
  const float* A     = (const float*)d_in[1];
  const float* Wb    = (const float*)d_in[2];
  const float* bb    = (const float*)d_in[3];
  const float* Wc    = (const float*)d_in[4];
  const float* bc    = (const float*)d_in[5];
  const float* Wd    = (const float*)d_in[6];
  const float* bd    = (const float*)d_in[7];
  const float* gamma = (const float*)d_in[8];
  const float* beta  = (const float*)d_in[9];
  float* out = (float*)d_out;

  // Workspace layout (~86.4 MB)
  char* w = (char*)d_ws;
  float*    eM   = (float*)(w + 0x000000);
  float*    eT0  = (float*)(w + 0x040000);
  float*    eT1  = (float*)(w + 0x080000);
  float*    eP0  = (float*)(w + 0x0C0000);
  float*    eP1  = (float*)(w + 0x100000);
  _Float16* AdT  = (_Float16*)(w + 0x140000);   // [S,S] f16        128 KB
  _Float16* WbTh = (_Float16*)(w + 0x160000);   // [512,256] f16    256 KB
  _Float16* WcTh = (_Float16*)(w + 0x1A0000);   // [256,512] f16    256 KB
  _Float16* WdTh = (_Float16*)(w + 0x1E0000);   // [512,512] f16    512 KB
  _Float16* xh   = (_Float16*)(w + 0x260000);   // [M,512] f16       32 MB
  float*    Bxb  = (float*)(w + 0x2260000);     // [B,T,S] fp32      32 MB
  _Float16* hsb  = (_Float16*)(w + 0x4260000);  // [B,T,S] f16       16 MB

  // 0) one-time conversions / transposes to f16
  cvt_f16<<<(MROWS * HID) / 256, 256, 0, stream>>>(x, xh);
  transpose_f16<<<(ST * HID) / 256, 256, 0, stream>>>(Wb, WbTh, ST, HID);
  transpose_f16<<<(HID * ST) / 256, 256, 0, stream>>>(Wc, WcTh, HID, ST);
  transpose_f16<<<(HID * HID) / 256, 256, 0, stream>>>(Wd, WdTh, HID, HID);

  // 1) Ad = expm(A*DT); store Ad^T as f16
  expm_prep<<<256, 256, 0, stream>>>(A, eM, eT0, eP0, 0.1f / 16.f);
  float* tin = eT0; float* tout = eT1;
  for (int n = 2; n <= 12; ++n) {
    expm_mm<<<256, 256, 0, stream>>>(tin, eM, tout, eP0, 1.f / (float)n);
    float* tmp = tin; tin = tout; tout = tmp;
  }
  float* pin = eP0; float* pout = eP1;
  for (int s = 0; s < 4; ++s) {
    expm_mm<<<256, 256, 0, stream>>>(pin, pin, pout, nullptr, 1.f);
    float* tmp = pin; pin = pout; pout = tmp;
  }
  expm_transpose_f16<<<256, 256, 0, stream>>>(pin, AdT);

  // 2) Bx = x @ Wb^T + bb          [32768 x 256], K=512
  gemm_wmma_f16<false, false>
      <<<dim3(ST / 64, MROWS / 128), 256, 0, stream>>>(xh, WbTh, bb, nullptr,
                                                       Bxb, MROWS, ST, HID);

  // 3) recurrence -> hs (f16)
  ssm_scan<<<8, 512, 0, stream>>>(Bxb, AdT, hsb);

  // 4) out = x @ Wd^T + bd + x     [32768 x 512], K=512
  gemm_wmma_f16<false, true>
      <<<dim3(HID / 64, MROWS / 128), 256, 0, stream>>>(xh, WdTh, bd, x,
                                                        out, MROWS, HID, HID);

  // 5) out += hs @ Wc^T + bc       [32768 x 512], K=256
  gemm_wmma_f16<true, false>
      <<<dim3(HID / 64, MROWS / 128), 256, 0, stream>>>(hsb, WcTh, bc, nullptr,
                                                        out, MROWS, HID, ST);

  // 6) LayerNorm in place
  layernorm_512<<<MROWS, 256, 0, stream>>>(out, gamma, beta);
}
